// MultiHeadAttention_76244259439086
// MI455X (gfx1250) — matrix operand order repaired and checked by reading, not compiled, not run on hardware
//
#include <hip/hip_runtime.h>

// ---------------------------------------------------------------------------
// MultiHeadAttention forward for MI455X (gfx1250), wave32.
// bf16 WMMA everywhere, fp32 softmax/accumulators.
// Data movement: GEMM uses double-buffered GLOBAL_LOAD_ASYNC_TO_LDS_B128;
// attention K/V tiles use the Tensor Data Mover (TENSOR_LOAD_TO_LDS).
// B=2, S=2048, D=1024, H=16, Dh=64.
// ---------------------------------------------------------------------------

#define B_   2
#define S_   2048
#define D_   1024
#define H_   16
#define DH_  64
#define NTOK (B_ * S_)   // 4096

typedef __bf16 bf16;
typedef __attribute__((ext_vector_type(16))) __bf16 v16bf;
typedef __attribute__((ext_vector_type(8)))  __bf16 v8bf;
typedef __attribute__((ext_vector_type(4)))  __bf16 v4bf;
typedef __attribute__((ext_vector_type(8)))  float  v8f;
typedef __attribute__((ext_vector_type(4)))  int    v4i;
typedef __attribute__((ext_vector_type(4)))  unsigned int u32x4;
typedef __attribute__((ext_vector_type(8)))  int    i32x8;

// Pointer types for the async-to-LDS builtin: (global v4i*, lds v4i*, int, int).
typedef __attribute__((address_space(1))) v4i* gptr_v4i;
typedef __attribute__((address_space(3))) v4i* lptr_v4i;

#if defined(__has_builtin)
#  if __has_builtin(__builtin_amdgcn_tensor_load_to_lds)
#    define HAVE_TDM 1
#  endif
#endif
#ifndef HAVE_TDM
#  define HAVE_TDM 0
#endif

// ---------------------------------------------------------------------------
// Async global -> LDS copy of 16 bytes per lane (ASYNCcnt-tracked).
static __device__ __forceinline__ void async_copy_b128(bf16* lds_dst, const bf16* gsrc) {
#if defined(__has_builtin) && __has_builtin(__builtin_amdgcn_global_load_async_to_lds_b128)
    __builtin_amdgcn_global_load_async_to_lds_b128(
        (gptr_v4i)(unsigned long long)gsrc,
        (lptr_v4i)(unsigned)(unsigned long long)lds_dst, 0, 0);
#else
    unsigned l = (unsigned)(unsigned long long)lds_dst;
    unsigned long long g = (unsigned long long)gsrc;
    asm volatile("global_load_async_to_lds_b128 %0, %1, off"
                 :: "v"(l), "v"(g) : "memory");
#endif
}

static __device__ __forceinline__ void wait_async0() {
#if defined(__has_builtin) && __has_builtin(__builtin_amdgcn_s_wait_asynccnt)
    __builtin_amdgcn_s_wait_asynccnt(0);
#else
    asm volatile("s_wait_asynccnt 0x0" ::: "memory");
#endif
}

static __device__ __forceinline__ void wait_tensor0() {
#if defined(__has_builtin) && __has_builtin(__builtin_amdgcn_s_wait_tensorcnt)
    __builtin_amdgcn_s_wait_tensorcnt(0);
#else
    asm volatile("s_wait_tensorcnt 0x0" ::: "memory");
#endif
}

#if HAVE_TDM
// TDM 2D tile DMA: tile_d0 x tile_d1 elements of bf16, global row pitch
// `stride_elems`, packed densely into LDS at lds_dst.  D# layout per
// CDNA5 ISA (group0: count/lds/global/type,
// group1: data_size + tensor/tile dims + dim0 stride; groups 2-3 zero => 2D).
// This toolchain's builtin is the 6-arg form:
//   (uint32x4 g0, int32x8 g1, int32x4 g2, int32x4 g3, int32x8 extra, int cpol)
static __device__ __forceinline__ void tdm_load_2d(bf16* lds_dst, const bf16* gsrc,
                                                   int tile_d0, int tile_d1,
                                                   int stride_elems) {
    const unsigned long long ga = (unsigned long long)gsrc;
    const unsigned lds = (unsigned)(unsigned long long)lds_dst;
    u32x4 g0;
    g0[0] = 1u;                                              // count=1 (valid user D#)
    g0[1] = lds;                                             // lds_addr (bytes)
    g0[2] = (unsigned)(ga & 0xFFFFFFFFu);                    // global_addr[31:0]
    g0[3] = (unsigned)((ga >> 32) & 0x1FFFFFFu) | (2u << 30);// global_addr[56:32], type=2
    i32x8 g1;
    g1[0] = 1 << 16;                                         // data_size=1 (2 bytes)
    g1[1] = (tile_d0 & 0xFFFF) << 16;                        // tensor_dim0[15:0]
    g1[2] = ((unsigned)tile_d0 >> 16) | ((tile_d1 & 0xFFFF) << 16); // t_dim0 hi, t_dim1 lo
    g1[3] = ((unsigned)tile_d1 >> 16) | ((tile_d0 & 0xFFFF) << 16); // t_dim1 hi, tile_dim0
    g1[4] = tile_d1 & 0xFFFF;                                // tile_dim1 (tile_dim2=0)
    g1[5] = stride_elems;                                    // tensor_dim0_stride[31:0]
    g1[6] = 0;                                               // stride hi, dim1_stride lo
    g1[7] = 0;
    v4i  z4 = {0, 0, 0, 0};
    i32x8 z8 = {0, 0, 0, 0, 0, 0, 0, 0};
    __builtin_amdgcn_tensor_load_to_lds(g0, g1, z4, z4, z8, 0);
}
#endif

// ---------------------------------------------------------------------------
// Fragment loaders matching CDNA5 WMMA VGPR layouts (wave32).
// A (16x32 bf16): lane<16 -> row=lane, K={0..7,16..23}; lane>=16 -> K={8..15,24..31}.
static __device__ __forceinline__ v16bf load_a(const bf16* base, int ld, int lane) {
    const int row = lane & 15;
    const int kh  = lane >> 4;
    const bf16* p = base + row * ld + kh * 8;
    v8bf lo = *(const v8bf*)(p);
    v8bf hi = *(const v8bf*)(p + 16);
    v16bf a;
#pragma unroll
    for (int j = 0; j < 8; ++j) { a[j] = lo[j]; a[j + 8] = hi[j]; }
    return a;
}

// B (32x16 KxN): lane<16 -> col=lane, K={0..7,16..23}; lane>=16 -> col=lane-16, K={8..15,24..31}.
// baseT is N-major: baseT[n * ld + k] (column's K-run contiguous).
static __device__ __forceinline__ v16bf load_b(const bf16* baseT, int ld, int lane) {
    const int col = lane & 15;
    const int kh  = lane >> 4;
    const bf16* p = baseT + col * ld + kh * 8;
    v8bf lo = *(const v8bf*)(p);
    v8bf hi = *(const v8bf*)(p + 16);
    v16bf b;
#pragma unroll
    for (int j = 0; j < 8; ++j) { b[j] = lo[j]; b[j + 8] = hi[j]; }
    return b;
}

static __device__ __forceinline__ v8f wmma_bf16(v16bf a, v16bf b, v8f c) {
    return __builtin_amdgcn_wmma_f32_16x16x32_bf16(
        false, a, false, b, (short)0, c, false, false);
}

// ---------------------------------------------------------------------------
// fp32 -> bf16 conversion (4 elements / thread)
__global__ __launch_bounds__(256) void f32_to_bf16_vec4(const float* __restrict__ in,
                                                        bf16* __restrict__ out, int n4) {
    int i = blockIdx.x * blockDim.x + threadIdx.x;
    if (i < n4) {
        float4 f = ((const float4*)in)[i];
        v4bf o = { (bf16)f.x, (bf16)f.y, (bf16)f.z, (bf16)f.w };
        ((v4bf*)out)[i] = o;
    }
}

// fp32 [R x C] -> bf16 transposed [C x R] (weights: row-major KxN -> N-major).
__global__ __launch_bounds__(256)
void transpose_f32_to_bf16(const float* __restrict__ in, bf16* __restrict__ out,
                           int R, int C) {
    __shared__ float tl[32][33];
    const int c0 = blockIdx.x * 32;
    const int r0 = blockIdx.y * 32;
    const int tx = threadIdx.x & 31;
    const int ty = (threadIdx.x >> 5) * 4;
#pragma unroll
    for (int j = 0; j < 4; ++j)
        tl[ty + j][tx] = in[(size_t)(r0 + ty + j) * C + c0 + tx];
    __syncthreads();
#pragma unroll
    for (int j = 0; j < 4; ++j)
        out[(size_t)(c0 + ty + j) * R + r0 + tx] = (bf16)tl[tx][ty + j];
}

// bf16 batched transpose: per z, in [R x C] -> out [C x R]  (V -> Vt[d][s]).
__global__ __launch_bounds__(256)
void transpose_bf16_batched(const bf16* __restrict__ in, bf16* __restrict__ out,
                            int R, int C) {
    in  += (size_t)blockIdx.z * R * C;
    out += (size_t)blockIdx.z * R * C;
    __shared__ bf16 tl[32][33];
    const int c0 = blockIdx.x * 32;
    const int r0 = blockIdx.y * 32;
    const int tx = threadIdx.x & 31;
    const int ty = (threadIdx.x >> 5) * 4;
#pragma unroll
    for (int j = 0; j < 4; ++j)
        tl[ty + j][tx] = in[(size_t)(r0 + ty + j) * C + c0 + tx];
    __syncthreads();
#pragma unroll
    for (int j = 0; j < 4; ++j)
        out[(size_t)(c0 + ty + j) * R + r0 + tx] = tl[tx][ty + j];
}

// ---------------------------------------------------------------------------
// GEMM: C[M,N] = alpha * (A[M,K] @ W[K,N]) + bias[N]
// A bf16 row-major, Wt bf16 N-major (pre-transposed). 128x128 block tile,
// 8 waves, per-wave 32x64 (8 accumulators), double-buffered async LDS tiles.
static __device__ __forceinline__
void gemm_stage(const bf16* A, const bf16* Wt, bf16* As, bf16* Bs,
                int m0, int n0, int k0, int K, int tid) {
    const int r  = tid >> 1;            // 0..127
    const int ch = (tid & 1) * 16;      // 0 or 16
    const bf16* ga = &A[(size_t)(m0 + r) * K + k0 + ch];
    async_copy_b128(&As[r * 32 + ch],     ga);
    async_copy_b128(&As[r * 32 + ch + 8], ga + 8);
    const bf16* gb = &Wt[(size_t)(n0 + r) * K + k0 + ch];
    async_copy_b128(&Bs[r * 32 + ch],     gb);
    async_copy_b128(&Bs[r * 32 + ch + 8], gb + 8);
}

__global__ __launch_bounds__(256)
void gemm_bf16_wmma(const bf16* __restrict__ A, const bf16* __restrict__ Wt,
                    const float* __restrict__ bias, float alpha,
                    bf16* __restrict__ outB, float* __restrict__ outF,
                    int M, int K, int N) {
    __shared__ bf16 As[2][128 * 32];
    __shared__ bf16 Bs[2][128 * 32];

    const int tid  = threadIdx.x;
    const int lane = tid & 31;
    const int wave = tid >> 5;
    const int wm   = wave >> 1;            // 0..3  -> 32 rows
    const int wn   = wave & 1;             // 0..1  -> 64 cols
    const int m0   = blockIdx.y * 128;
    const int n0   = blockIdx.x * 128;

    v8f c[2][4];
#pragma unroll
    for (int i = 0; i < 2; ++i)
#pragma unroll
        for (int j = 0; j < 4; ++j) c[i][j] = v8f{};

    gemm_stage(A, Wt, As[0], Bs[0], m0, n0, 0, K, tid);
    int cur = 0;
    for (int k0 = 0; k0 < K; k0 += 32) {
        wait_async0();          // own async copies for buffer `cur` done
        __syncthreads();        // everyone's copies done; prev reads of cur^1 done
        if (k0 + 32 < K)
            gemm_stage(A, Wt, As[cur ^ 1], Bs[cur ^ 1], m0, n0, k0 + 32, K, tid);

        // Hoist all fragment loads into distinct registers, then burst 8 WMMAs.
        v16bf a0 = load_a(&As[cur][(wm * 32) * 32],      32, lane);
        v16bf a1 = load_a(&As[cur][(wm * 32 + 16) * 32], 32, lane);
        v16bf b0 = load_b(&Bs[cur][(wn * 64 +  0) * 32], 32, lane);
        v16bf b1 = load_b(&Bs[cur][(wn * 64 + 16) * 32], 32, lane);
        v16bf b2 = load_b(&Bs[cur][(wn * 64 + 32) * 32], 32, lane);
        v16bf b3 = load_b(&Bs[cur][(wn * 64 + 48) * 32], 32, lane);
        c[0][0] = wmma_bf16(a0, b0, c[0][0]);
        c[1][0] = wmma_bf16(a1, b0, c[1][0]);
        c[0][1] = wmma_bf16(a0, b1, c[0][1]);
        c[1][1] = wmma_bf16(a1, b1, c[1][1]);
        c[0][2] = wmma_bf16(a0, b2, c[0][2]);
        c[1][2] = wmma_bf16(a1, b2, c[1][2]);
        c[0][3] = wmma_bf16(a0, b3, c[0][3]);
        c[1][3] = wmma_bf16(a1, b3, c[1][3]);
        cur ^= 1;
    }

    const int lane15 = lane & 15;
    const int rhalf  = (lane >> 4) * 8;
#pragma unroll
    for (int i = 0; i < 2; ++i) {
        const int row0 = m0 + wm * 32 + i * 16 + rhalf;
#pragma unroll
        for (int j = 0; j < 4; ++j) {
            const int col = n0 + wn * 64 + j * 16 + lane15;
            const float bb = bias[col];
#pragma unroll
            for (int r = 0; r < 8; ++r) {
                const size_t idx = (size_t)(row0 + r) * N + col;
                float v = alpha * c[i][j][r] + bb;
                if (outF) outF[idx] = v;
                else      outB[idx] = (bf16)v;
            }
        }
    }
}

// ---------------------------------------------------------------------------
// Flash attention: per (b,h), 128 query rows per block (16/wave), KV tiled by
// 64 with online softmax. Scale folded into Q. K tile [key][d] and V tile
// (from pre-transposed Vt[d][s]) -> LDS [d][key], DMA'd by the TDM.
__global__ __launch_bounds__(256)
void flash_attn_wmma(const bf16* __restrict__ Q, const bf16* __restrict__ Kp,
                     const bf16* __restrict__ VtG, bf16* __restrict__ O) {
    __shared__ bf16 Kt[64 * 64];           // [key][d]
    __shared__ bf16 Vt[64 * 64];           // [d][key]
    __shared__ bf16 Pt[8][16 * 64];        // per-wave P tile [qrow][key]

    const int tid  = threadIdx.x;
    const int lane = tid & 31;
    const int wave = tid >> 5;
    const int h    = blockIdx.y;
    const int b    = blockIdx.z;
    const int q0   = blockIdx.x * 128 + wave * 16;

    const size_t qBase = (size_t)(b * S_ + q0) * D_ + h * DH_;
    v16bf aq0 = load_a(&Q[qBase],      D_, lane);
    v16bf aq1 = load_a(&Q[qBase + 32], D_, lane);

    float mrow[8], lrow[8];
    v8f oacc[4] = {v8f{}, v8f{}, v8f{}, v8f{}};
#pragma unroll
    for (int r = 0; r < 8; ++r) { mrow[r] = -1e30f; lrow[r] = 0.0f; }

    const int col   = lane & 15;
    const int rbase = (lane >> 4) * 8;

    for (int t0 = 0; t0 < S_; t0 += 64) {
        __syncthreads();   // all reads of Kt/Vt from previous tile are done
#if HAVE_TDM
        if (wave == 0) {
            // K tile: 64 keys x 64 d, row pitch D_;  V tile: 64 d x 64 keys, pitch S_.
            tdm_load_2d(Kt, &Kp[(size_t)(b * S_ + t0) * D_ + h * DH_], 64, 64, D_);
            tdm_load_2d(Vt, &VtG[(size_t)(b * D_ + h * DH_) * S_ + t0], 64, 64, S_);
            wait_tensor0();
        }
#else
        {
            const int row = tid >> 2;            // 0..63 (key for Kt, d for Vt)
            const int ch  = (tid & 3) * 16;      // 0,16,32,48
            const size_t gk = (size_t)(b * S_ + t0 + row) * D_ + h * DH_ + ch;
            async_copy_b128(&Kt[row * 64 + ch],     &Kp[gk]);
            async_copy_b128(&Kt[row * 64 + ch + 8], &Kp[gk + 8]);
            const size_t gv = (size_t)(b * D_ + h * DH_ + row) * S_ + t0 + ch;
            async_copy_b128(&Vt[row * 64 + ch],     &VtG[gv]);
            async_copy_b128(&Vt[row * 64 + ch + 8], &VtG[gv + 8]);
        }
        wait_async0();
#endif
        __syncthreads();

        // Scores: S = Q @ K^T (4 n-tiles of 16 keys, 2 K-steps over Dh).
        v8f sc[4];
#pragma unroll
        for (int nt = 0; nt < 4; ++nt) {
            v8f s = {};
            s = wmma_bf16(aq0, load_b(&Kt[nt * 16 * 64],      64, lane), s);
            s = wmma_bf16(aq1, load_b(&Kt[nt * 16 * 64 + 32], 64, lane), s);
            sc[nt] = s;
        }

        // Online softmax per row slot (reduce within 16-lane column group).
#pragma unroll
        for (int r = 0; r < 8; ++r) {
            float mt = fmaxf(fmaxf(sc[0][r], sc[1][r]), fmaxf(sc[2][r], sc[3][r]));
#pragma unroll
            for (int off = 8; off >= 1; off >>= 1)
                mt = fmaxf(mt, __shfl_xor(mt, off, 32));
            float mnew = fmaxf(mrow[r], mt);
            float corr = __expf(mrow[r] - mnew);
            float p0 = __expf(sc[0][r] - mnew);
            float p1 = __expf(sc[1][r] - mnew);
            float p2 = __expf(sc[2][r] - mnew);
            float p3 = __expf(sc[3][r] - mnew);
            float ps = (p0 + p1) + (p2 + p3);
#pragma unroll
            for (int off = 8; off >= 1; off >>= 1)
                ps += __shfl_xor(ps, off, 32);
            lrow[r] = lrow[r] * corr + ps;
            mrow[r] = mnew;
            oacc[0][r] *= corr; oacc[1][r] *= corr;
            oacc[2][r] *= corr; oacc[3][r] *= corr;
            const int prow = (rbase + r) * 64;
            Pt[wave][prow +  0 + col] = (bf16)p0;
            Pt[wave][prow + 16 + col] = (bf16)p1;
            Pt[wave][prow + 32 + col] = (bf16)p2;
            Pt[wave][prow + 48 + col] = (bf16)p3;
        }
        __syncthreads();   // P (C-layout) -> LDS -> A-layout reload

        // O += P @ V.
        v16bf ap0 = load_a(&Pt[wave][0],  64, lane);
        v16bf ap1 = load_a(&Pt[wave][32], 64, lane);
#pragma unroll
        for (int dnt = 0; dnt < 4; ++dnt) {
            oacc[dnt] = wmma_bf16(ap0, load_b(&Vt[dnt * 16 * 64],      64, lane), oacc[dnt]);
            oacc[dnt] = wmma_bf16(ap1, load_b(&Vt[dnt * 16 * 64 + 32], 64, lane), oacc[dnt]);
        }
    }

    // Normalize, store bf16 at [b, q, h*DH + d].
#pragma unroll
    for (int r = 0; r < 8; ++r) {
        const float inv = 1.0f / lrow[r];
        const size_t g = (size_t)(b * S_ + q0 + rbase + r) * D_ + h * DH_;
#pragma unroll
        for (int dnt = 0; dnt < 4; ++dnt)
            O[g + dnt * 16 + col] = (bf16)(oacc[dnt][r] * inv);
    }
}

// ---------------------------------------------------------------------------
extern "C" void kernel_launch(void* const* d_in, const int* in_sizes, int n_in,
                              void* d_out, int out_size, void* d_ws, size_t ws_size,
                              hipStream_t stream) {
    const float* x  = (const float*)d_in[0];
    const float* Wq = (const float*)d_in[1];
    const float* bq = (const float*)d_in[2];
    const float* Wk = (const float*)d_in[3];
    const float* bk = (const float*)d_in[4];
    const float* Wv = (const float*)d_in[5];
    const float* bv = (const float*)d_in[6];
    const float* Wo = (const float*)d_in[7];
    const float* bo = (const float*)d_in[8];
    float* out = (float*)d_out;

    // Workspace (bf16 staging), ~56 MB.
    char* ws = (char*)d_ws;
    bf16* Xb  = (bf16*)ws; ws += (size_t)NTOK * D_ * sizeof(bf16);
    bf16* Wqt = (bf16*)ws; ws += (size_t)D_ * D_ * sizeof(bf16);
    bf16* Wkt = (bf16*)ws; ws += (size_t)D_ * D_ * sizeof(bf16);
    bf16* Wvt = (bf16*)ws; ws += (size_t)D_ * D_ * sizeof(bf16);
    bf16* Wot = (bf16*)ws; ws += (size_t)D_ * D_ * sizeof(bf16);
    bf16* Qb  = (bf16*)ws; ws += (size_t)NTOK * D_ * sizeof(bf16);
    bf16* Kb  = (bf16*)ws; ws += (size_t)NTOK * D_ * sizeof(bf16);
    bf16* Vb  = (bf16*)ws; ws += (size_t)NTOK * D_ * sizeof(bf16);
    bf16* Vt  = (bf16*)ws; ws += (size_t)NTOK * D_ * sizeof(bf16);
    bf16* Ab  = (bf16*)ws; ws += (size_t)NTOK * D_ * sizeof(bf16);

    // Conversions: x -> bf16; weights -> bf16 transposed (N-major).
    {
        int n4 = (NTOK * D_) / 4;
        f32_to_bf16_vec4<<<(n4 + 255) / 256, 256, 0, stream>>>(x, Xb, n4);
        dim3 tg(D_ / 32, D_ / 32);
        transpose_f32_to_bf16<<<tg, 256, 0, stream>>>(Wq, Wqt, D_, D_);
        transpose_f32_to_bf16<<<tg, 256, 0, stream>>>(Wk, Wkt, D_, D_);
        transpose_f32_to_bf16<<<tg, 256, 0, stream>>>(Wv, Wvt, D_, D_);
        transpose_f32_to_bf16<<<tg, 256, 0, stream>>>(Wo, Wot, D_, D_);
    }

    // Projections (scale 1/sqrt(64)=0.125 folded into Q).
    dim3 gg(D_ / 128, NTOK / 128);
    gemm_bf16_wmma<<<gg, 256, 0, stream>>>(Xb, Wqt, bq, 0.125f, Qb, nullptr, NTOK, D_, D_);
    gemm_bf16_wmma<<<gg, 256, 0, stream>>>(Xb, Wkt, bk, 1.0f,   Kb, nullptr, NTOK, D_, D_);
    gemm_bf16_wmma<<<gg, 256, 0, stream>>>(Xb, Wvt, bv, 1.0f,   Vb, nullptr, NTOK, D_, D_);

    // V -> Vt[d][s] per batch (amortized over all query tiles of each head).
    transpose_bf16_batched<<<dim3(D_ / 32, S_ / 32, B_), 256, 0, stream>>>(Vb, Vt, S_, D_);

    // Flash attention.
    flash_attn_wmma<<<dim3(S_ / 128, H_, B_), 256, 0, stream>>>(Qb, Kb, Vt, Ab);

    // Output projection to fp32.
    gemm_bf16_wmma<<<gg, 256, 0, stream>>>(Ab, Wot, bo, 1.0f, nullptr, out, NTOK, D_, D_);
}
